// GraphTransformer_31318901522650
// MI455X (gfx1250) — matrix operand order repaired and checked by reading, not compiled, not run on hardware
//
#include <hip/hip_runtime.h>
#include <hip/hip_bf16.h>

typedef __attribute__((ext_vector_type(16))) _Float16 v16h;
typedef __attribute__((ext_vector_type(8)))  float    v8f;

#define HHEADS 8
#define HDIM   16
#define HID    128

// ---------------------------------------------------------------------------
// Pack W[K_DIM, N_DIM] (f32, row-major) into WMMA B-fragment order (f16),
// zero-padding K to a multiple of 32. Fragment layout (CDNA5 ISA 7.12.2,
// B 32x16 f16): element e of lane's v16h = W[kc + kbase + e + (e<8 ? 0 : 8)][t*16 + (lane&15)]
// with kbase = (lane<16 ? 0 : 8). Stored so each lane's 16 halves are a
// contiguous 32-byte run:  out[(((kcidx*NT)+t)*32 + lane)*16 + e]
// ---------------------------------------------------------------------------
template<int K_DIM, int N_DIM>
__global__ void pack_w_frag(const float* __restrict__ W, _Float16* __restrict__ out)
{
    constexpr int KP = (K_DIM + 31) & ~31;
    constexpr int NT = N_DIM / 16;
    int i = blockIdx.x * blockDim.x + threadIdx.x;
    if (i >= KP * N_DIM) return;
    int e     = i & 15;
    int lane  = (i >> 4) & 31;
    int rem   = i >> 9;
    int t     = rem % NT;
    int kcidx = rem / NT;
    int k = kcidx * 32 + ((lane < 16) ? 0 : 8) + ((e < 8) ? e : e + 8);
    int n = t * 16 + (lane & 15);
    float w = (k < K_DIM) ? W[k * N_DIM + n] : 0.0f;
    out[i] = (_Float16)w;
}

// ---------------------------------------------------------------------------
// WMMA GEMM: C[n_rows, N_DIM] = A[n_rows, K_DIM] @ W + bias (W pre-packed
// into fragment order, L2-resident). 256 threads = 8 wave32 waves; each wave
// computes a 16-row x N_DIM strip via v_wmma_f32_16x16x32_f16. No LDS.
//   A 16x32 f16 : lane<16 -> row=lane, K {0..7,16..23}; lane>=16 -> K {8..15,24..31}
//   C/D 16x16 f32: vgpr r -> row (lane<16 ? r : 8+r), col = lane&15
// ---------------------------------------------------------------------------
template<int K_DIM, int N_DIM, bool RELU>
__global__ __launch_bounds__(256) void wmma_gemm_bias(
    const float* __restrict__ A, const _Float16* __restrict__ Wf,
    const float* __restrict__ bias, float* __restrict__ C, int n_rows)
{
    constexpr int KP = (K_DIM + 31) & ~31;
    constexpr int NT = N_DIM / 16;
    constexpr int NKC = KP / 32;

    const int tid  = threadIdx.x;
    const int wave = tid >> 5;
    const int lane = tid & 31;
    const int l15  = lane & 15;
    const bool hi  = lane >= 16;

    const long row0 = ((long)blockIdx.x * 8 + wave) * 16;
    if (row0 >= n_rows) return;   // whole wave out of range -> uniform exit

    long arow = row0 + l15;
    if (arow >= n_rows) arow = n_rows - 1;   // clamp; stores masked later
    const float* Arow = A + arow * K_DIM;
    const int kbaseA = hi ? 8 : 0;

    const v16h* __restrict__ Wv = (const v16h*)Wf;

    v8f acc[NT];
    #pragma unroll
    for (int t = 0; t < NT; ++t) acc[t] = {};

    #pragma unroll
    for (int kcidx = 0; kcidx < NKC; ++kcidx) {
        const int kc = kcidx * 32;
        v16h af;
        if constexpr ((K_DIM & 31) == 0) {
            // aligned fast path: 4x float4 loads + packed cvt
            const float4* p0 = (const float4*)(Arow + kc + kbaseA);
            const float4* p1 = (const float4*)(Arow + kc + kbaseA + 16);
            float4 a0 = p0[0], a1 = p0[1], a2 = p1[0], a3 = p1[1];
            af[0]  = (_Float16)a0.x; af[1]  = (_Float16)a0.y;
            af[2]  = (_Float16)a0.z; af[3]  = (_Float16)a0.w;
            af[4]  = (_Float16)a1.x; af[5]  = (_Float16)a1.y;
            af[6]  = (_Float16)a1.z; af[7]  = (_Float16)a1.w;
            af[8]  = (_Float16)a2.x; af[9]  = (_Float16)a2.y;
            af[10] = (_Float16)a2.z; af[11] = (_Float16)a2.w;
            af[12] = (_Float16)a3.x; af[13] = (_Float16)a3.y;
            af[14] = (_Float16)a3.z; af[15] = (_Float16)a3.w;
        } else {
            // tail-guarded scalar path (input projection, K=165)
            #pragma unroll
            for (int e = 0; e < 8; ++e) {
                int k0 = kc + kbaseA + e;
                int k1 = kc + kbaseA + 16 + e;
                af[e]     = (_Float16)((k0 < K_DIM) ? Arow[k0] : 0.0f);
                af[8 + e] = (_Float16)((k1 < K_DIM) ? Arow[k1] : 0.0f);
            }
        }
        #pragma unroll
        for (int t = 0; t < NT; ++t) {
            v16h bf = Wv[(kcidx * NT + t) * 32 + lane];   // 2x global_load_b128
            acc[t] = __builtin_amdgcn_wmma_f32_16x16x32_f16(
                false, af, false, bf, (short)0, acc[t], false, false);
        }
    }

    #pragma unroll
    for (int t = 0; t < NT; ++t) {
        const int col = t * 16 + l15;
        const float bb = bias[col];
        #pragma unroll
        for (int r = 0; r < 8; ++r) {
            long grow = row0 + (hi ? 8 : 0) + r;
            if (grow < n_rows) {
                float v = acc[t][r] + bb;
                if (RELU) v = fmaxf(v, 0.0f);
                C[grow * N_DIM + col] = v;
            }
        }
    }
}

// ---------------------------------------------------------------------------
// utility / edge kernels
// ---------------------------------------------------------------------------
__global__ void fill_u32_kernel(unsigned int* p, unsigned int v, int n) {
    int i = blockIdx.x * blockDim.x + threadIdx.x;
    if (i < n) p[i] = v;
}

__device__ inline void atomicMaxF(float* addr, float value) {
    if (value >= 0.0f) atomicMax((int*)addr, __float_as_int(value));
    else               atomicMin((unsigned int*)addr, __float_as_uint(value));
}

// alpha[e,h] = dot(q[dst], k[src]) / sqrt(D); amax[d,h] = segment max
__global__ void edge_logits_max(const float* __restrict__ q,
                                const float* __restrict__ k,
                                const int* __restrict__ src,
                                const int* __restrict__ dst,
                                float* __restrict__ alpha,
                                float* __restrict__ amax, int ne)
{
    int idx = blockIdx.x * blockDim.x + threadIdx.x;
    if (idx >= ne * HHEADS) return;
    int e = idx / HHEADS, h = idx % HHEADS;
    int s = src[e], d = dst[e];
    const float4* qp = (const float4*)(q + (long)d * HID + h * HDIM);
    const float4* kp = (const float4*)(k + (long)s * HID + h * HDIM);
    float sum = 0.0f;
    #pragma unroll
    for (int i = 0; i < 4; ++i) {
        float4 a = qp[i], b = kp[i];
        sum += a.x * b.x + a.y * b.y + a.z * b.z + a.w * b.w;
    }
    sum *= 0.25f;   // 1/sqrt(16)
    alpha[idx] = sum;
    atomicMaxF(&amax[d * HHEADS + h], sum);
}

// ex[e,h] = exp(alpha - amax[dst]); denom[d,h] += ex
__global__ void edge_exp_sum(const float* __restrict__ alpha,
                             const float* __restrict__ amax,
                             const int* __restrict__ dst,
                             float* __restrict__ ex,
                             float* __restrict__ denom, int ne)
{
    int idx = blockIdx.x * blockDim.x + threadIdx.x;
    if (idx >= ne * HHEADS) return;
    int e = idx / HHEADS, h = idx % HHEADS;
    int d = dst[e];
    float v = expf(alpha[idx] - amax[d * HHEADS + h]);
    ex[idx] = v;
    atomicAdd(&denom[d * HHEADS + h], v);
}

// msg[dst, h, :] += v[src, h, :] * ex/(denom+eps)
__global__ void edge_aggregate(const float* __restrict__ ex,
                               const float* __restrict__ denom,
                               const float* __restrict__ vv,
                               const int* __restrict__ src,
                               const int* __restrict__ dst,
                               float* __restrict__ msg, int ne)
{
    int idx = blockIdx.x * blockDim.x + threadIdx.x;
    if (idx >= ne * HHEADS) return;
    int e = idx / HHEADS, h = idx % HHEADS;
    int s = src[e], d = dst[e];
    float a = ex[idx] / (denom[d * HHEADS + h] + 1e-16f);
    const float* vp = vv + (long)s * HID + h * HDIM;
    float* mp = msg + (long)d * HID + h * HDIM;
    #pragma unroll
    for (int i = 0; i < HDIM; ++i) atomicAdd(&mp[i], vp[i] * a);
}

// h = LayerNorm(h + msg + skip) * g + b, one wave32 per node
__global__ __launch_bounds__(256) void residual_ln(
    float* __restrict__ h, const float* __restrict__ msg,
    const float* __restrict__ skip, const float* __restrict__ g,
    const float* __restrict__ b, int n)
{
    int node = (blockIdx.x * blockDim.x + threadIdx.x) >> 5;
    int lane = threadIdx.x & 31;
    if (node >= n) return;
    float*       hp = h    + (long)node * HID + lane * 4;
    const float* mp = msg  + (long)node * HID + lane * 4;
    const float* sp = skip + (long)node * HID + lane * 4;

    float4 hv = *(const float4*)hp;
    float4 mv = *(const float4*)mp;
    float4 sv = *(const float4*)sp;
    float vals[4] = { hv.x + mv.x + sv.x, hv.y + mv.y + sv.y,
                      hv.z + mv.z + sv.z, hv.w + mv.w + sv.w };
    float sum = vals[0] + vals[1] + vals[2] + vals[3];
    #pragma unroll
    for (int off = 16; off > 0; off >>= 1) sum += __shfl_xor(sum, off, 32);
    float mean = sum * (1.0f / HID);
    float vs = 0.0f;
    #pragma unroll
    for (int i = 0; i < 4; ++i) { float d = vals[i] - mean; vs += d * d; }
    #pragma unroll
    for (int off = 16; off > 0; off >>= 1) vs += __shfl_xor(vs, off, 32);
    float rstd = rsqrtf(vs * (1.0f / HID) + 1e-5f);

    const float4 gv = *(const float4*)(g + lane * 4);
    const float4 bv = *(const float4*)(b + lane * 4);
    float4 res;
    res.x = (vals[0] - mean) * rstd * gv.x + bv.x;
    res.y = (vals[1] - mean) * rstd * gv.y + bv.y;
    res.z = (vals[2] - mean) * rstd * gv.z + bv.z;
    res.w = (vals[3] - mean) * rstd * gv.w + bv.w;
    *(float4*)hp = res;
}

// out[n,2] = a[n,64] @ Wc2[64,2] + bc2  (tiny tail GEMM, scalar)
__global__ void classifier_tail(const float* __restrict__ a,
                                const float* __restrict__ W,
                                const float* __restrict__ bias,
                                float* __restrict__ out, int n)
{
    int i = blockIdx.x * blockDim.x + threadIdx.x;
    if (i >= n) return;
    const float* ap = a + (long)i * 64;
    float o0 = bias[0], o1 = bias[1];
    #pragma unroll 8
    for (int k = 0; k < 64; ++k) {
        float av = ap[k];
        o0 += av * W[k * 2 + 0];
        o1 += av * W[k * 2 + 1];
    }
    out[(long)i * 2 + 0] = o0;
    out[(long)i * 2 + 1] = o1;
}

// ---------------------------------------------------------------------------
extern "C" void kernel_launch(void* const* d_in, const int* in_sizes, int n_in,
                              void* d_out, int out_size, void* d_ws, size_t ws_size,
                              hipStream_t stream)
{
    const float* x    = (const float*)d_in[0];
    const int*   ei   = (const int*)d_in[1];
    const float* W_in = (const float*)d_in[2];
    const float* b_in = (const float*)d_in[3];
    // layer 1: 4..13, layer 2: 14..23, classifier: 24..27
    const float* Wc1 = (const float*)d_in[24];
    const float* bc1 = (const float*)d_in[25];
    const float* Wc2 = (const float*)d_in[26];
    const float* bc2 = (const float*)d_in[27];

    const int n  = in_sizes[0] / 165;
    const int ne = in_sizes[1] / 2;
    const int* src = ei;
    const int* dst = ei + ne;

    // workspace carve-out
    char* ws = (char*)d_ws;
    size_t off = 0;
    auto carve = [&](size_t bytes) -> void* {
        void* p = ws + off;
        off += (bytes + 255) & ~(size_t)255;
        return p;
    };
    const size_t nodeMat = (size_t)n * HID * sizeof(float);
    float* h     = (float*)carve(nodeMat);
    float* bq    = (float*)carve(nodeMat);
    float* bk    = (float*)carve(nodeMat);
    float* bv    = (float*)carve(nodeMat);
    float* bs    = (float*)carve(nodeMat);
    float* bmsg  = (float*)carve(nodeMat);
    float* alpha = (float*)carve((size_t)ne * HHEADS * sizeof(float));
    float* exb   = (float*)carve((size_t)ne * HHEADS * sizeof(float));
    float* amax  = (float*)carve((size_t)n * HHEADS * sizeof(float));
    float* denom = (float*)carve((size_t)n * HHEADS * sizeof(float));
    float* c1    = (float*)carve((size_t)n * 64 * sizeof(float));
    // packed f16 weight fragments (L2-resident)
    _Float16* fWin = (_Float16*)carve((size_t)192 * 128 * 2);
    _Float16* fWq  = (_Float16*)carve((size_t)128 * 128 * 2);
    _Float16* fWk  = (_Float16*)carve((size_t)128 * 128 * 2);
    _Float16* fWv  = (_Float16*)carve((size_t)128 * 128 * 2);
    _Float16* fWs  = (_Float16*)carve((size_t)128 * 128 * 2);
    _Float16* fWc1 = (_Float16*)carve((size_t)128 * 64 * 2);

    const dim3 gemmGrid((n + 127) / 128);
    const int edgeTot = ne * HHEADS;
    const dim3 edgeGrid((edgeTot + 255) / 256);
    const dim3 nodeWaveGrid(((size_t)n * 32 + 255) / 256);
    const dim3 nodeGrid((n + 255) / 256);
    const int nhFill  = n * HHEADS;
    const int msgFill = n * HID;

    // input projection: h = x @ W_in + b_in
    pack_w_frag<165, 128><<<(192 * 128 + 255) / 256, 256, 0, stream>>>(W_in, fWin);
    wmma_gemm_bias<165, 128, false><<<gemmGrid, 256, 0, stream>>>(x, fWin, b_in, h, n);

    auto run_layer = [&](int base) {
        const float* Wq  = (const float*)d_in[base + 0];
        const float* bqv = (const float*)d_in[base + 1];
        const float* Wk  = (const float*)d_in[base + 2];
        const float* bkv = (const float*)d_in[base + 3];
        const float* Wv  = (const float*)d_in[base + 4];
        const float* bvv = (const float*)d_in[base + 5];
        const float* Ws  = (const float*)d_in[base + 6];
        const float* bsv = (const float*)d_in[base + 7];
        const float* g   = (const float*)d_in[base + 8];
        const float* be  = (const float*)d_in[base + 9];

        const dim3 pg((128 * 128 + 255) / 256);
        pack_w_frag<128, 128><<<pg, 256, 0, stream>>>(Wq, fWq);
        pack_w_frag<128, 128><<<pg, 256, 0, stream>>>(Wk, fWk);
        pack_w_frag<128, 128><<<pg, 256, 0, stream>>>(Wv, fWv);
        pack_w_frag<128, 128><<<pg, 256, 0, stream>>>(Ws, fWs);

        wmma_gemm_bias<128, 128, false><<<gemmGrid, 256, 0, stream>>>(h, fWq, bqv, bq, n);
        wmma_gemm_bias<128, 128, false><<<gemmGrid, 256, 0, stream>>>(h, fWk, bkv, bk, n);
        wmma_gemm_bias<128, 128, false><<<gemmGrid, 256, 0, stream>>>(h, fWv, bvv, bv, n);
        wmma_gemm_bias<128, 128, false><<<gemmGrid, 256, 0, stream>>>(h, fWs, bsv, bs, n);

        fill_u32_kernel<<<(nhFill + 255) / 256, 256, 0, stream>>>(
            (unsigned int*)amax, 0xFF800000u, nhFill);               // -inf
        fill_u32_kernel<<<(nhFill + 255) / 256, 256, 0, stream>>>(
            (unsigned int*)denom, 0u, nhFill);
        fill_u32_kernel<<<(msgFill + 255) / 256, 256, 0, stream>>>(
            (unsigned int*)bmsg, 0u, msgFill);

        edge_logits_max<<<edgeGrid, 256, 0, stream>>>(bq, bk, src, dst, alpha, amax, ne);
        edge_exp_sum<<<edgeGrid, 256, 0, stream>>>(alpha, amax, dst, exb, denom, ne);
        edge_aggregate<<<edgeGrid, 256, 0, stream>>>(exb, denom, bv, src, dst, bmsg, ne);

        residual_ln<<<nodeWaveGrid, 256, 0, stream>>>(h, bmsg, bs, g, be, n);
    };

    run_layer(4);   // layer 1
    run_layer(14);  // layer 2

    // classifier: relu(h @ Wc1 + bc1) @ Wc2 + bc2
    pack_w_frag<128, 64><<<(128 * 64 + 255) / 256, 256, 0, stream>>>(Wc1, fWc1);
    wmma_gemm_bias<128, 64, true><<<gemmGrid, 256, 0, stream>>>(h, fWc1, bc1, c1, n);
    classifier_tail<<<nodeGrid, 256, 0, stream>>>(c1, Wc2, bc2, (float*)d_out, n);
}
